// TemporalAttention_38843684225597
// MI455X (gfx1250) — compile-verified, matrix operand order checked by reading, not run on hardware
//
#include <hip/hip_runtime.h>
#include <hip/hip_bf16.h>

static constexpr int Bk = 16;
static constexpr int Ck = 128;
static constexpr int Lk = 2048;
static constexpr float LOG2E = 1.4426950408889634f;

typedef __attribute__((ext_vector_type(16))) __bf16 v16bf_t;
typedef __attribute__((ext_vector_type(8)))  __bf16 bf16x8;
typedef __attribute__((ext_vector_type(8)))  float  v8f;

union Frag16 { v16bf_t v; bf16x8 h[2]; };

// ---- DPP 16-lane butterfly max (row boundaries == fragment halves) --------
template<int CTRL>
__device__ __forceinline__ float dpp_mov(float x) {
    return __int_as_float(__builtin_amdgcn_update_dpp(
        0, __float_as_int(x), CTRL, 0xf, 0xf, true));
}
__device__ __forceinline__ float rowmax16(float x) {
    x = fmaxf(x, dpp_mov<0x140>(x)); // row_mirror       : xor 15
    x = fmaxf(x, dpp_mov<0x141>(x)); // row_half_mirror  : xor 7
    x = fmaxf(x, dpp_mov<0x1B>(x));  // quad_perm(3,2,1,0): xor 3
    x = fmaxf(x, dpp_mov<0xB1>(x));  // quad_perm(1,0,3,2): xor 1
    return x;
}

// ---------------------------------------------------------------------------
// Kernel 0: convert the three 128x128 weight matrices to bf16.
// ---------------------------------------------------------------------------
__global__ __launch_bounds__(256) void cvt_w(
    const float* __restrict__ W1, const float* __restrict__ W2,
    const float* __restrict__ W3, __bf16* __restrict__ Wb)
{
    const int i = blockIdx.x * 256 + threadIdx.x;
    Wb[i]               = (__bf16)W1[i];
    Wb[Ck * Ck + i]     = (__bf16)W2[i];
    Wb[2 * Ck * Ck + i] = (__bf16)W3[i];
}

// ---------------------------------------------------------------------------
// Kernel 1: LDS-tiled transpose-convert x[B,C,L] fp32 -> XT[B,L,C] bf16.
// ---------------------------------------------------------------------------
__global__ __launch_bounds__(256) void transpose_x(
    const float* __restrict__ x, __bf16* __restrict__ XT)
{
    __shared__ __bf16 t[32][33];
    const int b  = blockIdx.z;
    const int l0 = blockIdx.x * 32;
    const int c0 = blockIdx.y * 32;
    const int tx = threadIdx.x & 31;
    const int ty = threadIdx.x >> 5;   // 0..7
    #pragma unroll
    for (int k = 0; k < 4; ++k) {
        const int c = c0 + ty + k * 8;
        t[ty + k * 8][tx] = (__bf16)x[((size_t)b * Ck + c) * Lk + l0 + tx];
    }
    __syncthreads();
    #pragma unroll
    for (int k = 0; k < 4; ++k) {
        const int l = l0 + ty + k * 8;
        XT[((size_t)b * Lk + l) * Ck + c0 + tx] = t[tx][ty + k * 8];
    }
}

// ---------------------------------------------------------------------------
// Kernel 2: WMMA QKV projection. One wave owns 16 positions. The same
// register image serves as A- and B-fragment (layouts are symmetric):
//   Q,K:  D[o,l] = W  (A) x XT (B)  -> lane-contiguous b128 stores to [L,C]
//   V:    D[l,o] = XT (A) x W3 (B)  -> lane-contiguous b128 stores to [C,L]
// Q is pre-scaled by log2(e) so flash attention runs on raw v_exp_f32.
// ---------------------------------------------------------------------------
__global__ __launch_bounds__(32) void proj_wmma(
    const __bf16* __restrict__ XT, const __bf16* __restrict__ Wb,
    const float* __restrict__ b1, const float* __restrict__ b2,
    const float* __restrict__ b3,
    __bf16* __restrict__ Qo, __bf16* __restrict__ Ko, __bf16* __restrict__ Vo)
{
    const int b    = blockIdx.y;
    const int i0   = blockIdx.x * 16;
    const int ln   = threadIdx.x & 31;
    const int l16  = ln & 15;
    const int half = ln >> 4;
    const int koff = half * 8;

    const __bf16* Xb = XT + (size_t)b * Lk * Ck;

    Frag16 xa[4];                       // X tile, rows i0..i0+15
    #pragma unroll
    for (int kc = 0; kc < 4; ++kc) {
        const __bf16* p = Xb + (size_t)(i0 + l16) * Ck + kc * 32 + koff;
        xa[kc].h[0] = *(const bf16x8*)p;
        xa[kc].h[1] = *(const bf16x8*)(p + 16);
    }

    #pragma unroll
    for (int mat = 0; mat < 3; ++mat) {
        const __bf16* Wm   = Wb + (size_t)mat * Ck * Ck;
        const float*  bias = (mat == 0) ? b1 : (mat == 1) ? b2 : b3;
        const float   oscl = (mat == 0) ? LOG2E : 1.0f;
        #pragma unroll 1
        for (int ot = 0; ot < 8; ++ot) {
            Frag16 wf[4];               // W tile, rows ot*16..+15
            #pragma unroll
            for (int kc = 0; kc < 4; ++kc) {
                const __bf16* p = Wm + (size_t)(ot * 16 + l16) * Ck + kc * 32 + koff;
                wf[kc].h[0] = *(const bf16x8*)p;
                wf[kc].h[1] = *(const bf16x8*)(p + 16);
            }
            v8f acc = {0,0,0,0,0,0,0,0};
            if (mat < 2) {
                // D[o, l] = W x XT   (A=W, B=XT)
                #pragma unroll
                for (int kc = 0; kc < 4; ++kc)
                    acc = __builtin_amdgcn_wmma_f32_16x16x32_bf16(
                        false, wf[kc].v, false, xa[kc].v, (short)0, acc, false, false);
                // o = ot*16 + r + 8*half (elements), l = i0 + l16 (lane)
                bf16x8 hv;
                #pragma unroll
                for (int r = 0; r < 8; ++r)
                    hv[r] = (__bf16)((acc[r] + bias[ot * 16 + 8 * half + r]) * oscl);
                __bf16* dst = (mat == 0 ? Qo : Ko)
                            + ((size_t)b * Lk + i0 + l16) * Ck + ot * 16 + 8 * half;
                *(bf16x8*)dst = hv;
            } else {
                // D[l, o] = XT x W3^T (A=XT, B=W3)
                #pragma unroll
                for (int kc = 0; kc < 4; ++kc)
                    acc = __builtin_amdgcn_wmma_f32_16x16x32_bf16(
                        false, xa[kc].v, false, wf[kc].v, (short)0, acc, false, false);
                // l = i0 + r + 8*half (elements), o = ot*16 + l16 (lane)
                const float bv = bias[ot * 16 + l16];
                bf16x8 hv;
                #pragma unroll
                for (int r = 0; r < 8; ++r)
                    hv[r] = (__bf16)(acc[r] + bv);
                __bf16* dst = Vo + ((size_t)b * Ck + ot * 16 + l16) * Lk + i0 + 8 * half;
                *(bf16x8*)dst = hv;
            }
        }
    }
}

// ---------------------------------------------------------------------------
// Kernel 3: fused flash attention. One wave32 owns a 32-row Q block (two
// 16-row tiles), streaming 32-column K/V tiles: each K/V B-fragment feeds two
// WMMAs (halves L2 fragment traffic). Row max via DPP butterflies, softmax
// denominator via WMMA against an all-ones fragment, exp2-domain softmax,
// and a wave-uniform lazy-rescale branch (EXEC stays all-ones for WMMA).
// ---------------------------------------------------------------------------
__global__ __launch_bounds__(32) void attn_fused(
    const __bf16* __restrict__ Q, const __bf16* __restrict__ K,
    const __bf16* __restrict__ V, float* __restrict__ out)
{
    const int b    = blockIdx.y;
    const int i0   = blockIdx.x * 32;
    const int ln   = threadIdx.x & 31;
    const int l16  = ln & 15;
    const int half = ln >> 4;
    const int koff = half * 8;

    const __bf16* Qb = Q + (size_t)b * Lk * Ck;
    const __bf16* Kb = K + (size_t)b * Lk * Ck;
    const __bf16* Vb = V + (size_t)b * Ck * Lk;

    __shared__ float sP[2][16][36];     // D-layout -> A-layout transpose

    Frag16 qa[2][4];
    #pragma unroll
    for (int rt = 0; rt < 2; ++rt) {
        #pragma unroll
        for (int kc = 0; kc < 4; ++kc) {
            const __bf16* p = Qb + (size_t)(i0 + rt * 16 + l16) * Ck + kc * 32 + koff;
            qa[rt][kc].h[0] = *(const bf16x8*)p;
            qa[rt][kc].h[1] = *(const bf16x8*)(p + 16);
        }
    }

    Frag16 ones;                        // all-ones B-fragment for row sums
    #pragma unroll
    for (int e = 0; e < 8; ++e) { ones.h[0][e] = (__bf16)1.0f; ones.h[1][e] = (__bf16)1.0f; }

    v8f   o_acc[2][8];
    v8f   l_acc[2] = {{0,0,0,0,0,0,0,0}, {0,0,0,0,0,0,0,0}};
    float m_r[2][8];
    #pragma unroll
    for (int rt = 0; rt < 2; ++rt) {
        #pragma unroll
        for (int ch = 0; ch < 8; ++ch) {
            #pragma unroll
            for (int r = 0; r < 8; ++r) o_acc[rt][ch][r] = 0.0f;
        }
        #pragma unroll
        for (int r = 0; r < 8; ++r) m_r[rt][r] = -1e30f;
    }

    #pragma unroll 1
    for (int j0 = 0; j0 < Lk; j0 += 32) {
        // ---- S = Q . K^T, two 16x32 tiles; K fragments reused x2 ----
        v8f s[2][2];
        #pragma unroll
        for (int rt = 0; rt < 2; ++rt) {
            #pragma unroll
            for (int nt = 0; nt < 2; ++nt) {
                #pragma unroll
                for (int r = 0; r < 8; ++r) s[rt][nt][r] = 0.0f;
            }
        }
        #pragma unroll
        for (int kc = 0; kc < 4; ++kc) {
            Frag16 kb0, kb1;
            const __bf16* p0 = Kb + (size_t)(j0 + l16)      * Ck + kc * 32 + koff;
            const __bf16* p1 = Kb + (size_t)(j0 + 16 + l16) * Ck + kc * 32 + koff;
            kb0.h[0] = *(const bf16x8*)p0; kb0.h[1] = *(const bf16x8*)(p0 + 16);
            kb1.h[0] = *(const bf16x8*)p1; kb1.h[1] = *(const bf16x8*)(p1 + 16);
            #pragma unroll
            for (int rt = 0; rt < 2; ++rt) {
                s[rt][0] = __builtin_amdgcn_wmma_f32_16x16x32_bf16(
                    false, qa[rt][kc].v, false, kb0.v, (short)0, s[rt][0], false, false);
                s[rt][1] = __builtin_amdgcn_wmma_f32_16x16x32_bf16(
                    false, qa[rt][kc].v, false, kb1.v, (short)0, s[rt][1], false, false);
            }
        }

        // ---- online softmax in exp2 domain, row max via DPP ----
        float scale[2][8];
        bool  changed = false;
        #pragma unroll
        for (int rt = 0; rt < 2; ++rt) {
            #pragma unroll
            for (int r = 0; r < 8; ++r) {
                const float t    = rowmax16(fmaxf(s[rt][0][r], s[rt][1][r]));
                const float mnew = fmaxf(m_r[rt][r], t);
                scale[rt][r] = exp2f(m_r[rt][r] - mnew);
                changed     |= (scale[rt][r] < 1.0f);
                m_r[rt][r]   = mnew;
            }
        }

        __syncthreads();
        #pragma unroll
        for (int rt = 0; rt < 2; ++rt) {
            #pragma unroll
            for (int r = 0; r < 8; ++r) {
                sP[rt][r + 8 * half][l16]      = exp2f(s[rt][0][r] - m_r[rt][r]);
                sP[rt][r + 8 * half][16 + l16] = exp2f(s[rt][1][r] - m_r[rt][r]);
            }
        }
        __syncthreads();

        // ---- reload P tiles as A-fragments (16x32), convert to bf16 ----
        Frag16 pa[2];
        #pragma unroll
        for (int rt = 0; rt < 2; ++rt) {
            const float4 f0 = *(const float4*)&sP[rt][l16][koff];
            const float4 f1 = *(const float4*)&sP[rt][l16][koff + 4];
            const float4 f2 = *(const float4*)&sP[rt][l16][koff + 16];
            const float4 f3 = *(const float4*)&sP[rt][l16][koff + 20];
            bf16x8 h0, h1;
            h0[0]=(__bf16)f0.x; h0[1]=(__bf16)f0.y; h0[2]=(__bf16)f0.z; h0[3]=(__bf16)f0.w;
            h0[4]=(__bf16)f1.x; h0[5]=(__bf16)f1.y; h0[6]=(__bf16)f1.z; h0[7]=(__bf16)f1.w;
            h1[0]=(__bf16)f2.x; h1[1]=(__bf16)f2.y; h1[2]=(__bf16)f2.z; h1[3]=(__bf16)f2.w;
            h1[4]=(__bf16)f3.x; h1[5]=(__bf16)f3.y; h1[6]=(__bf16)f3.z; h1[7]=(__bf16)f3.w;
            pa[rt].h[0] = h0; pa[rt].h[1] = h1;
        }

        // ---- lazy rescale: wave-uniform branch, usually skipped ----
        if (__ballot(changed) != 0ull) {
            #pragma unroll
            for (int rt = 0; rt < 2; ++rt) {
                #pragma unroll
                for (int r = 0; r < 8; ++r) l_acc[rt][r] *= scale[rt][r];
                #pragma unroll
                for (int ch = 0; ch < 8; ++ch) {
                    #pragma unroll
                    for (int r = 0; r < 8; ++r) o_acc[rt][ch][r] *= scale[rt][r];
                }
            }
        }

        // ---- denom += P.1 ; O += P.V (V fragments reused x2) ----
        #pragma unroll
        for (int rt = 0; rt < 2; ++rt)
            l_acc[rt] = __builtin_amdgcn_wmma_f32_16x16x32_bf16(
                false, pa[rt].v, false, ones.v, (short)0, l_acc[rt], false, false);
        #pragma unroll
        for (int ch = 0; ch < 8; ++ch) {
            Frag16 vb;
            const __bf16* p = Vb + (size_t)(16 * ch + l16) * Lk + j0 + koff;
            vb.h[0] = *(const bf16x8*)p;
            vb.h[1] = *(const bf16x8*)(p + 16);
            #pragma unroll
            for (int rt = 0; rt < 2; ++rt)
                o_acc[rt][ch] = __builtin_amdgcn_wmma_f32_16x16x32_bf16(
                    false, pa[rt].v, false, vb.v, (short)0, o_acc[rt][ch], false, false);
        }
    }

    // ---- epilogue: divide by denom (broadcast in all lanes), store fp32 ----
    #pragma unroll
    for (int rt = 0; rt < 2; ++rt) {
        #pragma unroll
        for (int ch = 0; ch < 8; ++ch) {
            float4 lo, hi;
            lo.x = o_acc[rt][ch][0] / l_acc[rt][0]; lo.y = o_acc[rt][ch][1] / l_acc[rt][1];
            lo.z = o_acc[rt][ch][2] / l_acc[rt][2]; lo.w = o_acc[rt][ch][3] / l_acc[rt][3];
            hi.x = o_acc[rt][ch][4] / l_acc[rt][4]; hi.y = o_acc[rt][ch][5] / l_acc[rt][5];
            hi.z = o_acc[rt][ch][6] / l_acc[rt][6]; hi.w = o_acc[rt][ch][7] / l_acc[rt][7];
            float* op = out + ((size_t)b * Ck + 16 * ch + l16) * Lk
                            + i0 + rt * 16 + 8 * half;
            *(float4*)op       = lo;
            *(float4*)(op + 4) = hi;
        }
    }
}

// ---------------------------------------------------------------------------
extern "C" void kernel_launch(void* const* d_in, const int* in_sizes, int n_in,
                              void* d_out, int out_size, void* d_ws, size_t ws_size,
                              hipStream_t stream)
{
    const float* x  = (const float*)d_in[0];
    const float* W1 = (const float*)d_in[1];
    const float* b1 = (const float*)d_in[2];
    const float* W2 = (const float*)d_in[3];
    const float* b2 = (const float*)d_in[4];
    const float* W3 = (const float*)d_in[5];
    const float* b3 = (const float*)d_in[6];
    float* out = (float*)d_out;

    const size_t nQKV = (size_t)Bk * Lk * Ck;      // 4 Mi elements
    __bf16* Qw  = (__bf16*)d_ws;                   // 8 MB
    __bf16* Kw  = Qw + nQKV;                       // 8 MB
    __bf16* Vw  = Kw + nQKV;                       // 8 MB ([B,C,L] transposed)
    __bf16* XTw = Vw + nQKV;                       // 8 MB ([B,L,C] bf16)
    __bf16* Wbw = XTw + nQKV;                      // 96 KB (3 x 128 x 128)

    cvt_w      <<<Ck * Ck / 256, 256, 0, stream>>>(W1, W2, W3, Wbw);
    transpose_x<<<dim3(Lk / 32, Ck / 32, Bk), 256, 0, stream>>>(x, XTw);
    proj_wmma  <<<dim3(Lk / 16, Bk), 32, 0, stream>>>(XTw, Wbw, b1, b2, b3,
                                                      Qw, Kw, Vw);
    attn_fused <<<dim3(Lk / 32, Bk), 32, 0, stream>>>(Qw, Kw, Vw, out);
}
